// ProductQuantizer_23330262352613
// MI455X (gfx1250) — compile-verified, matrix operand order checked by reading, not run on hardware
//
#include <hip/hip_runtime.h>

// ---------------------------------------------------------------------------
// Product quantizer (Gumbel-softmax VQ) forward, fully fused for MI455X.
// B=16 T=2048 H=1024  G=2 V=320 D=128  TAU=1
//   q   [B,T,H]  -> d_out[0 .. 33554431]
//   perplexity   -> d_out[33554432]
//   commit_loss  -> d_out[33554433]
// All GEMMs run on v_wmma_f32_16x16x32_bf16 (wave32, f32 accumulate).
// d_ws layout (bytes):
//   [0      .. 4095 ]  stats: 641 f32 (prob column sums + commit-loss sum)
//   [4096   ..      ]  W_in   bf16 [640][1024]   (row-major)
//   [+1310720       ]  W_out  bf16 [1024][256]   (row-major)
//   [+524288        ]  codebook^T bf16 [2][128][320]  (d-major for contiguous K)
// total ~2.0 MB of workspace.
// ---------------------------------------------------------------------------

typedef __attribute__((ext_vector_type(16))) __bf16 v16bf;
typedef __attribute__((ext_vector_type(8)))  float  v8f;

union BFrag { v16bf v; uint4 q2[2]; unsigned short u[16]; };

__device__ __forceinline__ unsigned short f2bf(float f) {
    unsigned int x = __float_as_uint(f);
    x += 0x7fffu + ((x >> 16) & 1u);        // round-to-nearest-even
    return (unsigned short)(x >> 16);
}

__device__ __forceinline__ v8f zero8() {
    v8f z = {0.f, 0.f, 0.f, 0.f, 0.f, 0.f, 0.f, 0.f};
    return z;
}

#define WMMA_BF16(A, B, C) \
    __builtin_amdgcn_wmma_f32_16x16x32_bf16(false, (A), false, (B), (short)0, (C), false, false)

#define N_WIN  (640 * 1024)
#define N_WOUT (1024 * 256)
#define N_CB   (2 * 128 * 320)

// ---- Prep: zero stats; convert weights fp32->bf16; transpose codebook -----
__global__ __launch_bounds__(256) void pq_prep_kernel(
    const float* __restrict__ W_in, const float* __restrict__ W_out,
    const float* __restrict__ codebook, float* __restrict__ stats,
    unsigned short* __restrict__ winb, unsigned short* __restrict__ woutb,
    unsigned short* __restrict__ cbtb)
{
    const int tid = blockIdx.x * 256 + threadIdx.x;
    const int np  = gridDim.x * 256;
    for (int i = tid; i < 641; i += np) stats[i] = 0.f;
    for (int i = tid; i < N_WIN; i += np)  winb[i]  = f2bf(W_in[i]);
    for (int i = tid; i < N_WOUT; i += np) woutb[i] = f2bf(W_out[i]);
    for (int i = tid; i < N_CB; i += np) {             // cbT[g][d][v] = cb[g][v][d]
        const int g = i / (128 * 320);
        const int r = i - g * 128 * 320;
        const int d = r / 320;
        const int v = r - d * 320;
        cbtb[i] = f2bf(codebook[((size_t)g * 320 + v) * 128 + d]);
    }
}

__global__ __launch_bounds__(256) void pq_fused_kernel(
    const float* __restrict__ x,              // [32768,1024]
    const float* __restrict__ b_in,           // [640]
    const float* __restrict__ b_out,          // [1024]
    const float* __restrict__ gumbel_u,       // [32768,640]
    const unsigned short* __restrict__ winb,  // bf16 [640][1024]
    const unsigned short* __restrict__ woutb, // bf16 [1024][256]
    const unsigned short* __restrict__ cbtb,  // bf16 [2][128][320]
    float* __restrict__ q_out,                // [32768,1024]
    float* __restrict__ stats)
{
    // regionA is time-multiplexed: x tile (bf16) -> logits/probs (f32) -> q_mid (bf16)
    __shared__ __align__(16) unsigned char  regionA[41472];
    __shared__ __align__(16) unsigned short p_bf[16 * 648];  // probs as bf16
    __shared__ float wsum[8];

    unsigned short* xs  = (unsigned short*)regionA;  // [16][1032] bf16 x tile
    float*          lgt = (float*)regionA;           // [16][648]  f32 logits/probs
    unsigned short* qm  = (unsigned short*)regionA;  // [16][264]  bf16 q_mid

    const int tid  = threadIdx.x;
    const int wv   = tid >> 5;       // wave id 0..7
    const int lane = tid & 31;
    const int hlf  = lane >> 4;      // lane-half (wave32 WMMA layout)
    const int m    = lane & 15;      // A row / B column within a 16-tile
    const int ko   = hlf * 8;        // K sub-offset per lane half
    const size_t tok0 = (size_t)blockIdx.x * 16;

    // ---- Phase 1: stage x tile (fp32 -> packed bf16) into LDS -------------
    for (int i = tid; i < 16 * 512; i += 256) {
        const int r = i >> 9, c2 = i & 511;
        const float2 xv = *(const float2*)(x + (tok0 + r) * 1024 + c2 * 2);
        const unsigned int pk = ((unsigned int)f2bf(xv.y) << 16) | f2bf(xv.x);
        ((unsigned int*)xs)[r * 516 + c2] = pk;      // row stride 1032 ushorts
    }
    __syncthreads();

    // ---- Phase 2: logits = x @ W_in^T   (16 x 640, K = 1024) --------------
    v8f acc[5];
#pragma unroll
    for (int t = 0; t < 5; ++t) acc[t] = zero8();

    for (int kb = 0; kb < 32; ++kb) {
        const int K = kb * 32;
        BFrag a;
        a.q2[0] = *(const uint4*)(xs + m * 1032 + K + ko);       // K 0..7 / 8..15
        a.q2[1] = *(const uint4*)(xs + m * 1032 + K + 16 + ko);  // K 16..23 / 24..31
#pragma unroll
        for (int t = 0; t < 5; ++t) {
            const int n = (wv * 5 + t) * 16 + m;                 // W_in row = logit col
            const unsigned short* wr = winb + (size_t)n * 1024 + K + ko;
            BFrag b;
            b.q2[0] = *(const uint4*)(wr);
            b.q2[1] = *(const uint4*)(wr + 16);
            acc[t] = WMMA_BF16(a.v, b.v, acc[t]);
        }
    }
    __syncthreads();                       // xs dead -> regionA becomes lgt

#pragma unroll
    for (int t = 0; t < 5; ++t) {
        const int n = (wv * 5 + t) * 16 + m;
        const float bv = b_in[n];
#pragma unroll
        for (int r = 0; r < 8; ++r)        // C/D layout: rows r + 8*hlf, col = n
            lgt[(r + hlf * 8) * 648 + n] = acc[t][r] + bv;
    }
    __syncthreads();

    // ---- Phase 3: Gumbel-softmax per (token, group) row of 320 ------------
#pragma unroll
    for (int qr = 0; qr < 4; ++qr) {
        const int rr = wv * 4 + qr;        // 32 rows: 16 tokens x 2 groups
        const int mr = rr >> 1, gg = rr & 1;
        float*          row  = lgt  + mr * 648 + gg * 320;
        unsigned short* prow = p_bf + mr * 648 + gg * 320;
        const float*    urow = gumbel_u + (tok0 + mr) * 640 + gg * 320;
        float z[10];
        float mx = -3.402823e38f;
#pragma unroll
        for (int j = 0; j < 10; ++j) {
            const int c = lane + j * 32;
            const float gmb = -__logf(-__logf(urow[c]));
            z[j] = row[c] + gmb;           // TAU == 1
            mx = fmaxf(mx, z[j]);
        }
#pragma unroll
        for (int off = 16; off > 0; off >>= 1) mx = fmaxf(mx, __shfl_xor(mx, off, 32));
        float s = 0.f;
#pragma unroll
        for (int j = 0; j < 10; ++j) { z[j] = __expf(z[j] - mx); s += z[j]; }
#pragma unroll
        for (int off = 16; off > 0; off >>= 1) s += __shfl_xor(s, off, 32);
        const float inv = 1.f / s;
#pragma unroll
        for (int j = 0; j < 10; ++j) {
            const int c = lane + j * 32;
            const float pv = z[j] * inv;
            row[c]  = pv;
            prow[c] = f2bf(pv);
        }
    }
    __syncthreads();

    // ---- Phase 4: per-column prob sums (for perplexity) -------------------
    for (int c = tid; c < 640; c += 256) {
        float s = 0.f;
#pragma unroll
        for (int r = 0; r < 16; ++r) s += lgt[r * 648 + c];
        atomicAdd(&stats[c], s);
    }
    __syncthreads();                       // lgt dead -> regionA becomes qm

    // ---- Phase 5: q_mid = probs @ codebook   (16 x 256, K = 320/group) ----
    const int g5 = wv >> 2;                // waves 0-3 -> group 0, 4-7 -> group 1
    v8f qacc[2];
    qacc[0] = zero8(); qacc[1] = zero8();
    for (int kb = 0; kb < 10; ++kb) {
        const int K = kb * 32;
        BFrag a;
        a.q2[0] = *(const uint4*)(p_bf + m * 648 + g5 * 320 + K + ko);
        a.q2[1] = *(const uint4*)(p_bf + m * 648 + g5 * 320 + K + 16 + ko);
#pragma unroll
        for (int tt = 0; tt < 2; ++tt) {
            const int n = (wv * 2 + tt) * 16 + m;   // column in [0,256)
            const int d = n & 127;
            const unsigned short* cr = cbtb + ((size_t)(g5 * 128 + d)) * 320 + K + ko;
            BFrag b;
            b.q2[0] = *(const uint4*)(cr);
            b.q2[1] = *(const uint4*)(cr + 16);
            qacc[tt] = WMMA_BF16(a.v, b.v, qacc[tt]);
        }
    }
#pragma unroll
    for (int tt = 0; tt < 2; ++tt) {
        const int n = (wv * 2 + tt) * 16 + m;
#pragma unroll
        for (int r = 0; r < 8; ++r)
            qm[(r + hlf * 8) * 264 + n] = f2bf(qacc[tt][r]);
    }
    __syncthreads();

    // ---- Phase 6: q = q_mid @ W_out^T + b_out  (16 x 1024, K = 256) -------
    v8f oacc[8];
#pragma unroll
    for (int i = 0; i < 8; ++i) oacc[i] = zero8();
    for (int kb = 0; kb < 8; ++kb) {
        const int K = kb * 32;
        BFrag a;
        a.q2[0] = *(const uint4*)(qm + m * 264 + K + ko);
        a.q2[1] = *(const uint4*)(qm + m * 264 + K + 16 + ko);
#pragma unroll
        for (int i = 0; i < 8; ++i) {
            const int n = (wv * 8 + i) * 16 + m;    // output column in H
            const unsigned short* wr = woutb + (size_t)n * 256 + K + ko;
            BFrag b;
            b.q2[0] = *(const uint4*)(wr);
            b.q2[1] = *(const uint4*)(wr + 16);
            oacc[i] = WMMA_BF16(a.v, b.v, oacc[i]);
        }
    }

    // epilogue: bias, store q, commit-loss partial
    float closs = 0.f;
#pragma unroll
    for (int i = 0; i < 8; ++i) {
        const int n = (wv * 8 + i) * 16 + m;
        const float bv = b_out[n];
#pragma unroll
        for (int r = 0; r < 8; ++r) {
            const size_t tok = tok0 + (size_t)(r + hlf * 8);
            const size_t idx = tok * 1024 + n;
            const float qv = oacc[i][r] + bv;
            q_out[idx] = qv;
            const float df = x[idx] - qv;
            closs += df * df;
        }
    }
#pragma unroll
    for (int off = 16; off > 0; off >>= 1) closs += __shfl_xor(closs, off, 32);
    if (lane == 0) wsum[wv] = closs;
    __syncthreads();
    if (tid == 0) {
        float s = 0.f;
#pragma unroll
        for (int w = 0; w < 8; ++w) s += wsum[w];
        atomicAdd(&stats[640], s);
    }
}

__global__ __launch_bounds__(32) void pq_finalize_kernel(const float* __restrict__ stats,
                                                         float* __restrict__ out)
{
    const int lane = threadIdx.x;
    float ent[2];
#pragma unroll
    for (int g = 0; g < 2; ++g) {
        float s = 0.f;
#pragma unroll
        for (int j = 0; j < 10; ++j) {
            const int c = lane + j * 32;
            const float p = stats[g * 320 + c] * (1.f / 32768.f);   // mean over B*T
            s += p * __logf(p + 1e-9f);
        }
#pragma unroll
        for (int off = 16; off > 0; off >>= 1) s += __shfl_xor(s, off, 32);
        ent[g] = -s;
    }
    if (lane == 0) {
        out[(size_t)33554432] = 0.5f * (__expf(ent[0]) + __expf(ent[1]));  // perplexity
        out[(size_t)33554433] = stats[640] * (1.f / 33554432.f);           // commit loss
    }
}

extern "C" void kernel_launch(void* const* d_in, const int* in_sizes, int n_in,
                              void* d_out, int out_size, void* d_ws, size_t ws_size,
                              hipStream_t stream)
{
    (void)in_sizes; (void)n_in; (void)out_size; (void)ws_size;
    const float* x    = (const float*)d_in[0];
    const float* Win  = (const float*)d_in[1];
    const float* bin  = (const float*)d_in[2];
    const float* cb   = (const float*)d_in[3];
    const float* Wout = (const float*)d_in[4];
    const float* bout = (const float*)d_in[5];
    const float* gu   = (const float*)d_in[6];
    float* out   = (float*)d_out;

    float* stats = (float*)d_ws;                                  // 641 f32
    unsigned short* winb  = (unsigned short*)d_ws + 2048;         // byte 4096
    unsigned short* woutb = winb + N_WIN;
    unsigned short* cbtb  = woutb + N_WOUT;

    pq_prep_kernel<<<1024, 256, 0, stream>>>(Win, Wout, cb, stats, winb, woutb, cbtb);
    pq_fused_kernel<<<2048, 256, 0, stream>>>(x, bin, bout, gu, winb, woutb, cbtb,
                                              out, stats);
    pq_finalize_kernel<<<1, 32, 0, stream>>>(stats, out);
}
